// MultiHeadSelfAttention_75788992906247
// MI455X (gfx1250) — matrix operand order repaired and checked by reading, not compiled, not run on hardware
//
#include <hip/hip_runtime.h>

// Problem constants (match reference)
#define B_  4
#define L_  2048
#define D_  1024
#define H_  16
#define HD_ 64
#define TD_ (3 * D_)   // 3072

typedef __attribute__((ext_vector_type(16))) _Float16 v16h;
typedef __attribute__((ext_vector_type(8)))  _Float16 v8h;
typedef __attribute__((ext_vector_type(8)))  float    v8f;
typedef __attribute__((ext_vector_type(4)))  unsigned int v4u;
typedef __attribute__((ext_vector_type(8)))  int      v8i_;
typedef __attribute__((ext_vector_type(4)))  int      v4i_;

union FragU { v16h v; v8h h[2]; };

// A-operand (16x32 f16, MxK) from row-major [M][ld] array (global or LDS).
// ISA layout: lanes 0-15 = M, halves 0-7 -> K 0..7, halves 8-15 -> K 16..23;
// lanes 16-31: K 8..15 / K 24..31.
static __device__ __forceinline__ v16h load_a_frag(const _Float16* base, int ld) {
  int lane = threadIdx.x & 31;
  int m  = lane & 15;
  int kh = (lane & 16) ? 8 : 0;
  const _Float16* p = base + m * ld + kh;
  FragU u;
  u.h[0] = *(const v8h*)(p);
  u.h[1] = *(const v8h*)(p + 16);
  return u.v;
}

// B-operand (32x16 f16, KxN) given B^T stored row-major [N][ld].
// ISA layout: lanes 0-15 (n=lane) hold K 0..7 / K 16..23, lanes 16-31 hold K 8..15 / K 24..31.
static __device__ __forceinline__ v16h load_b_frag(const _Float16* base, int ld) {
  int lane = threadIdx.x & 31;
  int n  = lane & 15;
  int kb = (lane & 16) ? 8 : 0;
  const _Float16* p = base + n * ld + kb;
  FragU u;
  u.h[0] = *(const v8h*)(p);
  u.h[1] = *(const v8h*)(p + 16);
  return u.v;
}

static __device__ __forceinline__ v8f wmma_f16(v16h a, v16h b, v8f c) {
  return __builtin_amdgcn_wmma_f32_16x16x32_f16(false, a, false, b, (short)0, c,
                                                false, false);
}

// ---------------- Tensor Data Mover: 2D tile Global -> LDS ----------------
// Builds a D# per CDNA5 ISA §8.3/8.4 and issues TENSOR_LOAD_TO_LDS.
// Tile: trows rows of tk f16 elements; global row stride = stride_elems.
// LDS dest: contiguous row-major [trows][tk] at byte offset lds_off.
static __device__ __forceinline__ void tdm_load_tile_2d(
    unsigned lds_off, const _Float16* g, int tk, int trows, long stride_elems) {
  unsigned long long ga = (unsigned long long)(uintptr_t)g;
  v4u g0;
  g0[0] = 1u;                                    // count=1, user descriptor
  g0[1] = lds_off;                               // lds_addr (bytes)
  g0[2] = (unsigned)(ga & 0xFFFFFFFFu);          // global_addr[31:0]
  g0[3] = (unsigned)((ga >> 32) & 0x01FFFFFFu)   // global_addr[56:32]
        | (2u << 30);                            // type = 2 ("image")
  long sb = stride_elems;
  v8i_ g1;
  g1[0] = (1 << 16);                             // data_size = 1 -> 2 bytes
  g1[1] = (int)((tk & 0xFFFF) << 16);            // tensor_dim0[15:0]
  g1[2] = (int)(((tk >> 16) & 0xFFFF) | ((trows & 0xFFFF) << 16)); // td0 hi | td1 lo
  g1[3] = (int)(((trows >> 16) & 0xFFFF) | ((tk & 0xFFFF) << 16)); // td1 hi | tile_dim0
  g1[4] = (int)(trows & 0xFFFF);                 // tile_dim1 (tile_dim2 = 0)
  g1[5] = (int)(sb & 0xFFFFFFFFL);               // tensor_dim0_stride[31:0]
  g1[6] = (int)((sb >> 32) & 0xFFFF);            // tensor_dim0_stride[47:32]
  g1[7] = 0;
  v4i_ g2 = {0, 0, 0, 0};
  v4i_ g3 = {0, 0, 0, 0};
  v8i_ g4 = {0, 0, 0, 0, 0, 0, 0, 0};            // extra group (6-arg toolchain)
  __builtin_amdgcn_tensor_load_to_lds(g0, g1, g2, g3, g4, 0);
}

static __device__ __forceinline__ unsigned lds_offset_of(const void* p) {
  return (unsigned)(uintptr_t)p;  // low 32 bits of generic ptr = LDS byte offset
}

// ---------------- conversion kernels ----------------

__global__ void cast_f32_to_f16_kernel(const float* __restrict__ in,
                                       _Float16* __restrict__ out, int n) {
  for (int i = blockIdx.x * blockDim.x + threadIdx.x; i < n;
       i += gridDim.x * blockDim.x)
    out[i] = (_Float16)in[i];
}

// in: row-major [K][N] f32 -> out: row-major [N][K] f16
__global__ void transpose_cast_kernel(const float* __restrict__ in,
                                      _Float16* __restrict__ out, int K, int N) {
  int total = K * N;
  for (int i = blockIdx.x * blockDim.x + threadIdx.x; i < total;
       i += gridDim.x * blockDim.x) {
    int n = i / K;
    int k = i - n * K;
    out[i] = (_Float16)in[k * N + n];
  }
}

// ---------------- tiled TDM GEMM core ----------------
// Block: 256 threads = 8 waves. Tile 128(M) x 128(N), K-step 32.
// Wave w: rows (w&3)*32..+31, cols (w>>2)*64..+63 -> acc[2][4] 16x16 tiles.
// Wave 0 drives the Tensor Data Mover; double-buffered LDS staging.
#define BM_ 128
#define BN_ 128
#define BK_ 32

struct GemmAcc { v8f acc[2][4]; };

static __device__ __forceinline__ void gemm_tile_tdm(
    const _Float16* __restrict__ A,  // [M][Kdim] row-major
    const _Float16* __restrict__ Bt, // [N][Kdim] row-major (B transposed)
    int Kdim, int m0, int n0, _Float16* As /*2*BM_*BK_*/,
    _Float16* Bs /*2*BM_*BK_*/, GemmAcc& r) {
  const int wid = threadIdx.x >> 5;
  const bool w0 = (threadIdx.x < 32);
  const int wm = wid & 3;
  const int wn = wid >> 2;

#pragma unroll
  for (int mt = 0; mt < 2; ++mt)
#pragma unroll
    for (int nt = 0; nt < 4; ++nt)
      r.acc[mt][nt] = (v8f)(0.0f);

  const int NK = Kdim / BK_;
  if (w0) {
    tdm_load_tile_2d(lds_offset_of(As), A + (size_t)m0 * Kdim, BK_, BM_, Kdim);
    tdm_load_tile_2d(lds_offset_of(Bs), Bt + (size_t)n0 * Kdim, BK_, BN_, Kdim);
  }
  int cur = 0;
  for (int kt = 0; kt < NK; ++kt) {
    if (w0) __builtin_amdgcn_s_wait_tensorcnt(0);
    __syncthreads();  // current buffer ready for all waves; prev reads done
    if (kt + 1 < NK && w0) {
      int nx = cur ^ 1;
      int k = (kt + 1) * BK_;
      tdm_load_tile_2d(lds_offset_of(As + nx * (BM_ * BK_)),
                       A + (size_t)m0 * Kdim + k, BK_, BM_, Kdim);
      tdm_load_tile_2d(lds_offset_of(Bs + nx * (BN_ * BK_)),
                       Bt + (size_t)n0 * Kdim + k, BK_, BN_, Kdim);
    }
    const _Float16* Ab = As + cur * (BM_ * BK_);
    const _Float16* Bb = Bs + cur * (BN_ * BK_);
    v16h af[2];
#pragma unroll
    for (int mt = 0; mt < 2; ++mt)
      af[mt] = load_a_frag(Ab + (wm * 32 + mt * 16) * BK_, BK_);
#pragma unroll
    for (int nt = 0; nt < 4; ++nt) {
      v16h bf = load_b_frag(Bb + (wn * 64 + nt * 16) * BK_, BK_);
#pragma unroll
      for (int mt = 0; mt < 2; ++mt)
        r.acc[mt][nt] = wmma_f16(af[mt], bf, r.acc[mt][nt]);
    }
    cur ^= 1;
  }
}

// ---------------- QKV projection GEMM ----------------
__global__ __launch_bounds__(256) void gemm_qkv_kernel(
    const _Float16* __restrict__ xh, const _Float16* __restrict__ wT,
    _Float16* __restrict__ Qs, _Float16* __restrict__ Ks,
    _Float16* __restrict__ Vt) {
  __shared__ __align__(16) _Float16 As[2 * BM_ * BK_];
  __shared__ __align__(16) _Float16 Bs[2 * BN_ * BK_];
  const int m0 = blockIdx.x * BM_;
  const int n0 = blockIdx.y * BN_;
  GemmAcc r;
  gemm_tile_tdm(xh, wT, D_, m0, n0, As, Bs, r);

  const int lane = threadIdx.x & 31;
  const int wid = threadIdx.x >> 5;
  const int wm = wid & 3, wn = wid >> 2;
  const int group = (lane >> 4) & 1;
  const int cn = lane & 15;
#pragma unroll
  for (int mt = 0; mt < 2; ++mt) {
#pragma unroll
    for (int nt = 0; nt < 4; ++nt) {
      int n = n0 + wn * 64 + nt * 16 + cn;
      int which = n / D_;
      int dcol  = n % D_;
      int h  = dcol / HD_;
      int hd = dcol % HD_;
#pragma unroll
      for (int j = 0; j < 8; ++j) {
        int m = m0 + wm * 32 + mt * 16 + j + group * 8;
        int b = m / L_;
        int l = m % L_;
        int bh = b * H_ + h;
        float val = r.acc[mt][nt][j];
        if (which == 0)
          Qs[(bh * L_ + l) * HD_ + hd] = (_Float16)(val * 0.125f); // 1/sqrt(64)
        else if (which == 1)
          Ks[(bh * L_ + l) * HD_ + hd] = (_Float16)val;
        else
          Vt[(bh * HD_ + hd) * L_ + l] = (_Float16)val;
      }
    }
  }
}

// ---------------- output projection GEMM ----------------
__global__ __launch_bounds__(256) void gemm_proj_kernel(
    const _Float16* __restrict__ AO, const _Float16* __restrict__ wT,
    const float* __restrict__ bias, float* __restrict__ out) {
  __shared__ __align__(16) _Float16 As[2 * BM_ * BK_];
  __shared__ __align__(16) _Float16 Bs[2 * BN_ * BK_];
  const int m0 = blockIdx.x * BM_;
  const int n0 = blockIdx.y * BN_;
  GemmAcc r;
  gemm_tile_tdm(AO, wT, D_, m0, n0, As, Bs, r);

  const int lane = threadIdx.x & 31;
  const int wid = threadIdx.x >> 5;
  const int wm = wid & 3, wn = wid >> 2;
  const int group = (lane >> 4) & 1;
  const int cn = lane & 15;
#pragma unroll
  for (int mt = 0; mt < 2; ++mt) {
#pragma unroll
    for (int nt = 0; nt < 4; ++nt) {
      int n = n0 + wn * 64 + nt * 16 + cn;
      float bv = bias[n];
#pragma unroll
      for (int j = 0; j < 8; ++j) {
        int m = m0 + wm * 32 + mt * 16 + j + group * 8;
        out[(size_t)m * D_ + n] = r.acc[mt][nt][j] + bv;
      }
    }
  }
}

// ---------------- causal flash attention ----------------
// One wave32 per (b,h, 16-row q tile). Loops over 32-key chunks.
__global__ __launch_bounds__(32) void attn_kernel(
    const _Float16* __restrict__ Qs, const _Float16* __restrict__ Ks,
    const _Float16* __restrict__ Vt, _Float16* __restrict__ AO) {
  __shared__ __align__(16) _Float16 Plds[16 * 32];

  int lane = threadIdx.x & 31;
  int qt = blockIdx.x % (L_ / 16);
  int bh = blockIdx.x / (L_ / 16);
  int q0 = qt * 16;

  const _Float16* Qbase = Qs + (bh * L_ + q0) * HD_;
  const _Float16* Kbase = Ks + bh * L_ * HD_;
  const _Float16* Vbase = Vt + bh * HD_ * L_;

  v16h qa[2];
  qa[0] = load_a_frag(Qbase, HD_);
  qa[1] = load_a_frag(Qbase + 32, HD_);

  v8f o[4] = {};
  float mrow[8], lrow[8];
#pragma unroll
  for (int j = 0; j < 8; ++j) { mrow[j] = -1e30f; lrow[j] = 0.f; }

  int group = (lane >> 4) & 1;
  int cn = lane & 15;
  int nchunks = (q0 + 16 + 31) / 32;

  for (int c = 0; c < nchunks; ++c) {
    int k0 = c * 32;
    v8f s[2] = {};
#pragma unroll
    for (int ss = 0; ss < 2; ++ss) {
#pragma unroll
      for (int kk = 0; kk < 2; ++kk) {
        v16h kb = load_b_frag(Kbase + (k0 + ss * 16) * HD_ + kk * 32, HD_);
        s[ss] = wmma_f16(qa[kk], kb, s[ss]);
      }
    }
    // causal mask
#pragma unroll
    for (int ss = 0; ss < 2; ++ss) {
      int col = k0 + ss * 16 + cn;
#pragma unroll
      for (int j = 0; j < 8; ++j) {
        int row = q0 + j + group * 8;
        if (col > row) s[ss][j] = -1e30f;
      }
    }
    // online softmax: rows of C layout live in 16-lane halves -> xor reduce
    float alpha[8];
#pragma unroll
    for (int j = 0; j < 8; ++j) {
      float mx = fmaxf(s[0][j], s[1][j]);
#pragma unroll
      for (int off = 8; off; off >>= 1)
        mx = fmaxf(mx, __shfl_xor(mx, off, 32));
      float mnew = fmaxf(mrow[j], mx);
      float p0 = __expf(s[0][j] - mnew);
      float p1 = __expf(s[1][j] - mnew);
      s[0][j] = p0;
      s[1][j] = p1;
      float ps = p0 + p1;
#pragma unroll
      for (int off = 8; off; off >>= 1)
        ps += __shfl_xor(ps, off, 32);
      alpha[j] = __expf(mrow[j] - mnew);
      lrow[j] = lrow[j] * alpha[j] + ps;
      mrow[j] = mnew;
    }
#pragma unroll
    for (int nn = 0; nn < 4; ++nn)
#pragma unroll
      for (int j = 0; j < 8; ++j)
        o[nn][j] *= alpha[j];

    // P: C-layout -> LDS row-major [16][32] -> reload as A operand
#pragma unroll
    for (int ss = 0; ss < 2; ++ss)
#pragma unroll
      for (int j = 0; j < 8; ++j)
        Plds[(j + group * 8) * 32 + ss * 16 + cn] = (_Float16)s[ss][j];
    __syncthreads();
    v16h pa = load_a_frag(Plds, 32);
    __syncthreads();

    // O += P @ V  (V stored transposed [hd][L] => direct B operand)
#pragma unroll
    for (int nn = 0; nn < 4; ++nn) {
      v16h vb = load_b_frag(Vbase + (nn * 16) * L_ + k0, L_);
      o[nn] = wmma_f16(pa, vb, o[nn]);
    }
  }

  int b = bh / H_;
  int h = bh % H_;
#pragma unroll
  for (int j = 0; j < 8; ++j) {
    float inv = 1.0f / lrow[j];
    int row = b * L_ + q0 + j + group * 8;
#pragma unroll
    for (int nn = 0; nn < 4; ++nn)
      AO[(size_t)row * D_ + h * HD_ + nn * 16 + cn] = (_Float16)(o[nn][j] * inv);
  }
}

// ---------------- host launcher ----------------
extern "C" void kernel_launch(void* const* d_in, const int* in_sizes, int n_in,
                              void* d_out, int out_size, void* d_ws,
                              size_t ws_size, hipStream_t stream) {
  const float* x      = (const float*)d_in[0];
  // d_in[1] = attn_mask (causal, baked into the kernels)
  const float* w_qkv  = (const float*)d_in[2];
  const float* w_proj = (const float*)d_in[3];
  const float* b_proj = (const float*)d_in[4];
  float* out = (float*)d_out;

  const size_t N_X    = (size_t)B_ * L_ * D_;   // 8388608
  const size_t N_WQKV = (size_t)D_ * TD_;       // 3145728
  const size_t N_WPRJ = (size_t)D_ * D_;        // 1048576

  _Float16* p      = (_Float16*)d_ws;
  _Float16* xh     = p;            p += N_X;
  _Float16* wqkvT  = p;            p += N_WQKV;
  _Float16* wprojT = p;            p += N_WPRJ;
  _Float16* Qs     = p;            p += N_X;
  _Float16* Ks     = p;            p += N_X;
  _Float16* Vt     = p;            p += N_X;
  _Float16* AO     = p;            p += N_X;

  // 1) fp32 -> fp16 conversions (+ weight transposes for B-operand layout)
  cast_f32_to_f16_kernel<<<4096, 256, 0, stream>>>(x, xh, (int)N_X);
  transpose_cast_kernel<<<4096, 256, 0, stream>>>(w_qkv, wqkvT, D_, TD_);
  transpose_cast_kernel<<<2048, 256, 0, stream>>>(w_proj, wprojT, D_, D_);

  // 2) QKV projection: M=8192, N=3072, K=1024 ; 128x128 tiles
  gemm_qkv_kernel<<<dim3(8192 / BM_, TD_ / BN_), 256, 0, stream>>>(
      xh, wqkvT, Qs, Ks, Vt);

  // 3) causal flash attention: B*H*(L/16) = 8192 waves
  attn_kernel<<<dim3(B_ * H_ * (L_ / 16)), 32, 0, stream>>>(Qs, Ks, Vt, AO);

  // 4) output projection: M=8192, N=1024, K=1024 ; 128x128 tiles + bias
  gemm_proj_kernel<<<dim3(8192 / BM_, D_ / BN_), 256, 0, stream>>>(
      AO, wprojT, b_proj, out);
}
